// TransformerEncoderLayerWithAttn_52390011076733
// MI455X (gfx1250) — compile-verified, matrix operand order checked by reading, not run on hardware
//
#include <hip/hip_runtime.h>
#include <hip/hip_bf16.h>
#include <math.h>

// ---------------- problem constants ----------------
constexpr int kB   = 4;
constexpr int kL   = 1024;
constexpr int kD   = 1024;
constexpr int kH   = 16;
constexpr int kHD  = 64;       // head dim
constexpr int kF   = 4096;
constexpr int kLD3 = 3 * kD;   // qkv row stride
constexpr int kM   = kB * kL;  // 4096 token rows

typedef __bf16 bf16;
typedef __attribute__((ext_vector_type(8)))  bf16  v8bf;
typedef __attribute__((ext_vector_type(16))) bf16  v16bf;
typedef __attribute__((ext_vector_type(8)))  float v8f;

static __device__ __forceinline__ v8f wmma_bf16(v16bf a, v16bf b, v8f c) {
  // D = A(16x32 bf16) * B(32x16 bf16) + C(16x16 f32)
  return __builtin_amdgcn_wmma_f32_16x16x32_bf16(false, a, false, b,
                                                 (short)0, c, false, false);
}

static __device__ __forceinline__ v8bf cvt8(float4 a, float4 b) {
  v8bf r;
  r[0] = (bf16)a.x; r[1] = (bf16)a.y; r[2] = (bf16)a.z; r[3] = (bf16)a.w;
  r[4] = (bf16)b.x; r[5] = (bf16)b.y; r[6] = (bf16)b.z; r[7] = (bf16)b.w;
  return r;
}

// Load a 16(row)x32(K) bf16 fragment from LDS (row-major, pitch in elements,
// pitch % 8 == 0 so both 16B ds_load_b128 are aligned).
// Element layout per CDNA5 ISA: lanes 0-15 -> row=lane, K {0..7,16..23};
// lanes 16-31 -> row=lane-16, K {8..15,24..31}.
static __device__ __forceinline__ v16bf load_frag(const bf16* base, int pitch) {
  const int lane = threadIdx.x & 31;
  const bf16* p = base + (lane & 15) * pitch + (lane >> 4) * 8;
  v8bf lo = *(const v8bf*)p;
  v8bf hi = *(const v8bf*)(p + 16);
  return __builtin_shufflevector(lo, hi, 0, 1, 2, 3, 4, 5, 6, 7,
                                 8, 9, 10, 11, 12, 13, 14, 15);
}

// =====================================================================
// Generic GEMM: C[M,N] = A[M,K] * W[N,K]^T + bias[N]  (optional ReLU)
// Block tile 64x128, K-stage 64 -> 8 WMMAs per barrier pair.
// 8 waves, each wave owns a 32x32 patch (4 accumulator tiles).
// =====================================================================
constexpr int GBM = 64, GBN = 128, GBK = 64;

__global__ __launch_bounds__(256)
void gemm_bias_kernel(const float* __restrict__ A, const float* __restrict__ W,
                      const float* __restrict__ bias, float* __restrict__ C,
                      int M, int N, int K, int relu) {
  __shared__ bf16 sA[GBM][GBK];   // 8 KB
  __shared__ bf16 sW[GBN][GBK];   // 16 KB
  const int tid  = threadIdx.x;
  const int lane = tid & 31;
  const int wave = tid >> 5;
  const int rowb = blockIdx.y * GBM;
  const int colb = blockIdx.x * GBN;
  const int wm = (wave >> 2) * 32;  // wave row offset in block tile
  const int wn = (wave & 3) * 32;   // wave col offset in block tile

  const v8f vzero = {0.f, 0.f, 0.f, 0.f, 0.f, 0.f, 0.f, 0.f};
  v8f acc[2][2] = {{vzero, vzero}, {vzero, vzero}};

  const int ar = tid >> 2, ac = (tid & 3) * 16;  // A stage: 16 floats/thread
  const int wr = tid >> 1, wc = (tid & 1) * 32;  // W stage: 32 floats/thread

  for (int kk = 0; kk < K; kk += GBK) {
    {  // stage A 64x64 fp32 -> bf16
      const float* src = A + (size_t)(rowb + ar) * K + kk + ac;
      float4 f0 = *(const float4*)src;
      float4 f1 = *(const float4*)(src + 4);
      float4 f2 = *(const float4*)(src + 8);
      float4 f3 = *(const float4*)(src + 12);
      *(v8bf*)&sA[ar][ac]     = cvt8(f0, f1);
      *(v8bf*)&sA[ar][ac + 8] = cvt8(f2, f3);
      if (kk + GBK < K) __builtin_prefetch(src + GBK, 0, 3);  // WGP-scope
    }
    {  // stage W 128x64 fp32 -> bf16
      const float* src = W + (size_t)(colb + wr) * K + kk + wc;
#pragma unroll
      for (int j = 0; j < 32; j += 16) {
        float4 f0 = *(const float4*)(src + j);
        float4 f1 = *(const float4*)(src + j + 4);
        float4 f2 = *(const float4*)(src + j + 8);
        float4 f3 = *(const float4*)(src + j + 12);
        *(v8bf*)&sW[wr][wc + j]     = cvt8(f0, f1);
        *(v8bf*)&sW[wr][wc + j + 8] = cvt8(f2, f3);
      }
      if (kk + GBK < K) __builtin_prefetch(src + GBK, 0, 3);  // WGP-scope
    }
    __syncthreads();
#pragma unroll
    for (int ks = 0; ks < GBK; ks += 32) {
      v16bf aF0 = load_frag(&sA[wm][ks], GBK);
      v16bf aF1 = load_frag(&sA[wm + 16][ks], GBK);
      v16bf bF0 = load_frag(&sW[wn][ks], GBK);
      v16bf bF1 = load_frag(&sW[wn + 16][ks], GBK);
      acc[0][0] = wmma_bf16(aF0, bF0, acc[0][0]);
      acc[0][1] = wmma_bf16(aF0, bF1, acc[0][1]);
      acc[1][0] = wmma_bf16(aF1, bF0, acc[1][0]);
      acc[1][1] = wmma_bf16(aF1, bF1, acc[1][1]);
    }
    __syncthreads();
  }

  // Epilogue: C/D layout -> lane 0-15: n=lane, m=r; lane 16-31: n=lane-16, m=r+8
  const int half = lane >> 4;
#pragma unroll
  for (int mi = 0; mi < 2; ++mi) {
#pragma unroll
    for (int nj = 0; nj < 2; ++nj) {
      const int col = colb + wn + nj * 16 + (lane & 15);
      const float bv = bias[col];
#pragma unroll
      for (int r = 0; r < 8; ++r) {
        const int row = rowb + wm + mi * 16 + half * 8 + r;
        float v = acc[mi][nj][r] + bv;
        if (relu) v = fmaxf(v, 0.f);
        C[(size_t)row * N + col] = v;
      }
    }
  }
}

// =====================================================================
// Fused attention: one block per (b, h, 32 query rows).
// scores = softmax(Q K^T / 8 + mask) -> attn_w (global) ; ctx = P V
// =====================================================================
constexpr int QB       = 32;    // query rows per block
constexpr int SS_PITCH = 1028;  // fp32 score row pitch (padded)
constexpr int VT_PITCH = 136;   // bf16 V^T row pitch (17*8, 16B-aligned rows)
constexpr size_t SS_BYTES  = (size_t)QB * SS_PITCH * 4;   // 131584
constexpr size_t SP_BYTES  = (size_t)QB * kL * 2;         // 65536
constexpr size_t SQ_BYTES  = (size_t)QB * kHD * 2;        // 4096
constexpr size_t SK_BYTES  = (size_t)64 * kHD * 2;        // 8192
constexpr size_t SVT_BYTES = (size_t)kHD * VT_PITCH * 2;  // 17408
constexpr size_t ATTN_SMEM = SS_BYTES + SP_BYTES + SQ_BYTES + SK_BYTES + SVT_BYTES;

__global__ __launch_bounds__(256)
void attn_kernel(const float* __restrict__ qkv, const float* __restrict__ mask,
                 float* __restrict__ attw, float* __restrict__ ctx) {
  extern __shared__ char smem[];
  float* sS  = reinterpret_cast<float*>(smem);                                 // [QB][SS_PITCH]
  bf16*  sP  = reinterpret_cast<bf16*>(smem + SS_BYTES);                       // [QB][kL]
  bf16*  sQ  = reinterpret_cast<bf16*>(smem + SS_BYTES + SP_BYTES);            // [QB][kHD]
  bf16*  sK  = reinterpret_cast<bf16*>(smem + SS_BYTES + SP_BYTES + SQ_BYTES); // [64][kHD]
  bf16*  sVt = reinterpret_cast<bf16*>(smem + SS_BYTES + SP_BYTES + SQ_BYTES + SK_BYTES); // [kHD][VT_PITCH]

  const int tid = threadIdx.x, lane = tid & 31, wave = tid >> 5;
  const int half = lane >> 4;
  const int nQB = kL / QB;
  const int blk = blockIdx.x;
  const int b  = blk / (kH * nQB);
  const int h  = (blk / nQB) % kH;
  const int qb = blk % nQB;
  const size_t baseRow = (size_t)b * kL;
  const float* Qp = qkv + 0 * kD + h * kHD;
  const float* Kp = qkv + 1 * kD + h * kHD;
  const float* Vp = qkv + 2 * kD + h * kHD;

  const int mi = wave >> 2;  // 0..1 : 16-row tile within the 32 query rows
  const int nj = wave & 3;   // 0..3 : 16-col tile within each 64-wide chunk

  // ---- stage Q block [32][64] fp32 -> bf16 ----
  {
    const int r = tid >> 3, c = (tid & 7) * 8;
    const float* src = Qp + (baseRow + (size_t)qb * QB + r) * kLD3 + c;
    float4 f0 = *(const float4*)src;
    float4 f1 = *(const float4*)(src + 4);
    *(v8bf*)&sQ[r * kHD + c] = cvt8(f0, f1);
  }
  __syncthreads();
  const v16bf qF0 = load_frag(&sQ[(mi * 16) * kHD + 0], kHD);
  const v16bf qF1 = load_frag(&sQ[(mi * 16) * kHD + 32], kHD);

  // ---- scores: Q K^T * 0.125 + mask, 64 keys per stage ----
  const v8f vzero = {0.f, 0.f, 0.f, 0.f, 0.f, 0.f, 0.f, 0.f};
  for (int kc = 0; kc < kL; kc += 64) {
    {  // stage K tile [64 keys][64 d]
      const int r = tid >> 2, c = (tid & 3) * 16;
      const float* src = Kp + (baseRow + kc + r) * kLD3 + c;
      float4 f0 = *(const float4*)src;
      float4 f1 = *(const float4*)(src + 4);
      float4 f2 = *(const float4*)(src + 8);
      float4 f3 = *(const float4*)(src + 12);
      *(v8bf*)&sK[r * kHD + c]     = cvt8(f0, f1);
      *(v8bf*)&sK[r * kHD + c + 8] = cvt8(f2, f3);
    }
    __syncthreads();
    const v16bf kF0 = load_frag(&sK[(nj * 16) * kHD + 0], kHD);
    const v16bf kF1 = load_frag(&sK[(nj * 16) * kHD + 32], kHD);
    v8f s = vzero;
    s = wmma_bf16(qF0, kF0, s);
    s = wmma_bf16(qF1, kF1, s);
    const int kcCol = kc + nj * 16 + (lane & 15);
#pragma unroll
    for (int r = 0; r < 8; ++r) {
      const int lrow = mi * 16 + half * 8 + r;           // row within 32-q block
      const int qr = qb * QB + lrow;                     // global query row in [0,L)
      sS[lrow * SS_PITCH + kcCol] = s[r] * 0.125f + mask[(size_t)qr * kL + kcCol];
    }
    __syncthreads();
  }

  // ---- softmax per row (4 rows per wave, wave32 shuffles) ----
  for (int rr = 0; rr < 4; ++rr) {
    const int row = wave * 4 + rr;
    float* srow = &sS[row * SS_PITCH];
    float mx = -3.0e38f;
    for (int c = lane; c < kL; c += 32) mx = fmaxf(mx, srow[c]);
    for (int off = 16; off; off >>= 1) mx = fmaxf(mx, __shfl_xor(mx, off, 32));
    float sum = 0.f;
    for (int c = lane; c < kL; c += 32) {
      float p = __expf(srow[c] - mx);
      srow[c] = p;
      sum += p;
    }
    for (int off = 16; off; off >>= 1) sum += __shfl_xor(sum, off, 32);
    const float inv = 1.0f / sum;
    const size_t obase = (((size_t)b * kH + h) * kL + (qb * QB + row)) * kL;
    for (int c = lane; c < kL; c += 32) {
      float p = srow[c] * inv;
      attw[obase + c] = p;            // materialized attention weights
      sP[row * kL + c] = (bf16)p;     // bf16 copy for P*V
    }
  }
  __syncthreads();

  // ---- ctx = P[32,1024] * V[1024,64], stage V transposed 128 keys at a time ----
  v8f acc = vzero;
  for (int kc = 0; kc < kL; kc += 128) {
    {
      const int kp = tid >> 1, dbase = (tid & 1) * 32;
      const float* src = Vp + (baseRow + kc + kp) * kLD3 + dbase;
#pragma unroll
      for (int i = 0; i < 32; i += 4) {
        float4 f = *(const float4*)(src + i);
        sVt[(dbase + i + 0) * VT_PITCH + kp] = (bf16)f.x;
        sVt[(dbase + i + 1) * VT_PITCH + kp] = (bf16)f.y;
        sVt[(dbase + i + 2) * VT_PITCH + kp] = (bf16)f.z;
        sVt[(dbase + i + 3) * VT_PITCH + kp] = (bf16)f.w;
      }
    }
    __syncthreads();
#pragma unroll
    for (int ks = 0; ks < 4; ++ks) {
      v16bf pF = load_frag(&sP[(mi * 16) * kL + kc + ks * 32], kL);
      v16bf vF = load_frag(&sVt[(nj * 16) * VT_PITCH + ks * 32], VT_PITCH);
      acc = wmma_bf16(pF, vF, acc);
    }
    __syncthreads();
  }

  // write ctx tile: [b, l, h*64 + d]
  const int col = h * kHD + nj * 16 + (lane & 15);
  const int row0 = qb * QB + mi * 16 + half * 8;
#pragma unroll
  for (int r = 0; r < 8; ++r) {
    ctx[(baseRow + row0 + r) * kD + col] = acc[r];
  }
}

// =====================================================================
// y = LayerNorm(x + a) * g + b   (one block per token row, D=1024)
// =====================================================================
__global__ __launch_bounds__(256)
void add_ln_kernel(const float* __restrict__ x, const float* __restrict__ a,
                   const float* __restrict__ g, const float* __restrict__ bb,
                   float* __restrict__ out) {
  __shared__ float red[256];
  const int row = blockIdx.x;
  const int tid = threadIdx.x;
  const float* xr = x + (size_t)row * kD;
  const float* ar = a + (size_t)row * kD;
  float v[4];
  float s = 0.f;
#pragma unroll
  for (int i = 0; i < 4; ++i) {
    v[i] = xr[tid + i * 256] + ar[tid + i * 256];
    s += v[i];
  }
  red[tid] = s;
  __syncthreads();
  for (int off = 128; off; off >>= 1) {
    if (tid < off) red[tid] += red[tid + off];
    __syncthreads();
  }
  const float mu = red[0] * (1.0f / kD);
  __syncthreads();
  float sq = 0.f;
#pragma unroll
  for (int i = 0; i < 4; ++i) {
    float d = v[i] - mu;
    sq += d * d;
  }
  red[tid] = sq;
  __syncthreads();
  for (int off = 128; off; off >>= 1) {
    if (tid < off) red[tid] += red[tid + off];
    __syncthreads();
  }
  const float rstd = rsqrtf(red[0] * (1.0f / kD) + 1e-5f);
#pragma unroll
  for (int i = 0; i < 4; ++i) {
    const int c = tid + i * 256;
    out[(size_t)row * kD + c] = (v[i] - mu) * rstd * g[c] + bb[c];
  }
}

// =====================================================================
extern "C" void kernel_launch(void* const* d_in, const int* in_sizes, int n_in,
                              void* d_out, int out_size, void* d_ws, size_t ws_size,
                              hipStream_t stream) {
  (void)in_sizes; (void)n_in; (void)out_size; (void)ws_size;
  const float* x     = (const float*)d_in[0];
  const float* amask = (const float*)d_in[1];
  const float* in_w  = (const float*)d_in[2];
  const float* in_b  = (const float*)d_in[3];
  const float* op_w  = (const float*)d_in[4];
  const float* op_b  = (const float*)d_in[5];
  const float* l1w   = (const float*)d_in[6];
  const float* l1b   = (const float*)d_in[7];
  const float* l2w   = (const float*)d_in[8];
  const float* l2b   = (const float*)d_in[9];
  const float* g1    = (const float*)d_in[10];
  const float* b1    = (const float*)d_in[11];
  const float* g2    = (const float*)d_in[12];
  const float* b2    = (const float*)d_in[13];

  float* out_x  = (float*)d_out;                    // [B,L,D]
  float* out_aw = out_x + (size_t)kB * kL * kD;     // [B,H,L,L]

  float* ws       = (float*)d_ws;
  float* qkv      = ws;                                    // [4096,3072]
  float* ctx      = qkv + (size_t)kM * kLD3;               // [4096,1024]
  float* attn_out = ctx + (size_t)kM * kD;                 // [4096,1024]
  float* ff1      = attn_out + (size_t)kM * kD;            // [4096,4096]
  float* ff2      = ff1 + (size_t)kM * kF;                 // [4096,1024]

  // Allow >64KB dynamic LDS for the attention kernel (gfx1250 WGP has 320KB).
  (void)hipFuncSetAttribute((const void*)attn_kernel,
                            hipFuncAttributeMaxDynamicSharedMemorySize,
                            (int)ATTN_SMEM);

  // 1) qkv = x @ in_proj_w^T + in_proj_b
  gemm_bias_kernel<<<dim3(kLD3 / GBN, kM / GBM), 256, 0, stream>>>(
      x, in_w, in_b, qkv, kM, kLD3, kD, 0);
  // 2) attention: attn_w -> d_out, ctx -> ws
  attn_kernel<<<dim3(kB * kH * (kL / QB)), 256, ATTN_SMEM, stream>>>(
      qkv, amask, out_aw, ctx);
  // 3) attn_out = ctx @ out_proj_w^T + out_proj_b
  gemm_bias_kernel<<<dim3(kD / GBN, kM / GBM), 256, 0, stream>>>(
      ctx, op_w, op_b, attn_out, kM, kD, kD, 0);
  // 4) y1 = LN(x + attn_out)  (kept in the d_out x-region as scratch)
  add_ln_kernel<<<dim3(kM), 256, 0, stream>>>(x, attn_out, g1, b1, out_x);
  // 5) ff1 = relu(y1 @ lin1_w^T + lin1_b)
  gemm_bias_kernel<<<dim3(kF / GBN, kM / GBM), 256, 0, stream>>>(
      out_x, l1w, l1b, ff1, kM, kF, kD, 1);
  // 6) ff2 = ff1 @ lin2_w^T + lin2_b
  gemm_bias_kernel<<<dim3(kD / GBN, kM / GBM), 256, 0, stream>>>(
      ff1, l2w, l2b, ff2, kM, kD, kF, 0);
  // 7) out_x = LN(y1 + ff2)   (in-place safe: reads precede reduction barriers)
  add_ln_kernel<<<dim3(kM), 256, 0, stream>>>(out_x, ff2, g2, b2, out_x);
}
